// Deconv_2723009266544
// MI455X (gfx1250) — matrix operand order repaired and checked by reading, not compiled, not run on hardware
//
#include <hip/hip_runtime.h>
#include <hip/hip_bf16.h>

// Problem constants (match reference)
#define BB   8
#define CC   64
#define SRCN 16
#define HH   256
#define WW   256
#define HWN  (HH*WW)
#define EPSF 1e-16f

// Padded bf16 mirror geometry: 2-halo on all sides, width rounded so every
// row segment used by async-to-LDS is 16B aligned (264*2 = 528 = 33*16).
#define PH 260
#define PW 264
#define MIRSZ (PH*PW)

typedef __attribute__((ext_vector_type(16))) unsigned short v16u;
typedef __attribute__((ext_vector_type(16))) __bf16        v16bf;
typedef __attribute__((ext_vector_type(8)))  float         v8f;

#if defined(__AMDGCN__) && __has_builtin(__builtin_amdgcn_global_load_async_to_lds_b128)
#define HAS_ASYNC 1
#else
#define HAS_ASYNC 0
#endif

static __device__ __forceinline__ unsigned short f2bf(float f) {
  union { float f; unsigned u; } c; c.f = f;
  unsigned u = c.u;
  u += 0x7fffu + ((u >> 16) & 1u);   // round-to-nearest-even
  return (unsigned short)(u >> 16);
}

static __device__ __forceinline__ v8f wmma_bf16(const v16u& a, const v16u& b, v8f c) {
  return __builtin_amdgcn_wmma_f32_16x16x32_bf16(
      false, __builtin_bit_cast(v16bf, a),
      false, __builtin_bit_cast(v16bf, b),
      (short)0, c, false, false);
}

#if HAS_ASYNC
typedef __attribute__((ext_vector_type(4))) int v4i;
typedef __attribute__((address_space(1))) v4i gv4i;  // global 16B vector
typedef __attribute__((address_space(3))) v4i sv4i;  // LDS 16B vector
// Copy 16 bytes global->LDS via the async datapath (ASYNCcnt-tracked).
static __device__ __forceinline__ void async_cp16(const unsigned short* g, unsigned short* l) {
  __builtin_amdgcn_global_load_async_to_lds_b128((gv4i*)g, (sv4i*)l, 0, 0);
}
static __device__ __forceinline__ void wait_async0() {
#if __has_builtin(__builtin_amdgcn_s_wait_asynccnt)
  __builtin_amdgcn_s_wait_asynccnt(0);
#else
  asm volatile("s_wait_asynccnt 0" ::: "memory");
#endif
}
#endif

// ---------------------------------------------------------------------------
// s = W_lin @ x + b_lin   (per pixel; W_lin [16,64] cached in LDS)
// ---------------------------------------------------------------------------
__global__ __launch_bounds__(256) void init_s_kernel(
    const float* __restrict__ x, const float* __restrict__ W_lin,
    const float* __restrict__ b_lin, float* __restrict__ s) {
  __shared__ float wsh[SRCN * CC];
  __shared__ float bsh[SRCN];
  for (int i = threadIdx.x; i < SRCN * CC; i += blockDim.x) wsh[i] = W_lin[i];
  if (threadIdx.x < SRCN) bsh[threadIdx.x] = b_lin[threadIdx.x];
  __syncthreads();
  int idx = blockIdx.x * blockDim.x + threadIdx.x;
  if (idx >= BB * HWN) return;
  int b = idx / HWN, p = idx % HWN;
  float acc[SRCN];
#pragma unroll
  for (int o = 0; o < SRCN; ++o) acc[o] = bsh[o];
  for (int c = 0; c < CC; ++c) {
    float xv = x[((size_t)(b * CC + c)) * HWN + p];
#pragma unroll
    for (int o = 0; o < SRCN; ++o) acc[o] += wsh[o * CC + c] * xv;
  }
#pragma unroll
  for (int o = 0; o < SRCN; ++o) s[((size_t)(b * SRCN + o)) * HWN + p] = acc[o];
}

// f32 [img][H][W] -> zero-halo bf16 mirror [img][PH][PW]
__global__ void pack_pad_kernel(const float* __restrict__ src,
                                unsigned short* __restrict__ dst, int nimg) {
  int idx = blockIdx.x * blockDim.x + threadIdx.x;
  if (idx >= nimg * MIRSZ) return;
  int cx = idx % PW;
  int r = idx / PW;
  int cy = r % PH;
  int img = r / PH;
  int gy = cy - 2, gx = cx - 2;
  float v = 0.f;
  if (gy >= 0 && gy < HH && gx >= 0 && gx < WW)
    v = src[((size_t)img * HH + gy) * WW + gx];
  dst[idx] = f2bf(v);
}

__global__ void h_init_kernel(const float* __restrict__ h0, float* __restrict__ h, int n) {
  int idx = blockIdx.x * blockDim.x + threadIdx.x;
  if (idx < n) h[idx] = h0[idx % (CC * SRCN * 25)];
}

// Pack h (f32 [b][c][src][t]) into bf16 GEMM weight layouts:
//   hbf [b][c][kk = t*16 + src]        (conv(s, h),  O=64, I=16)
//   hTbf[b][src][kk = (24-t)*64 + c]   (conv(*, hT), O=16, I=64; flip folded)
__global__ void prep_weights_kernel(const float* __restrict__ h,
                                    unsigned short* __restrict__ hbf,
                                    unsigned short* __restrict__ hTbf, int n) {
  int idx = blockIdx.x * blockDim.x + threadIdx.x;
  if (idx >= n) return;
  int t = idx % 25;
  int r = idx / 25;
  int src = r % SRCN; r /= SRCN;
  int c = r % CC;
  int b = r / CC;
  unsigned short bv = f2bf(h[idx]);
  hbf[((size_t)(b * CC + c)) * (25 * SRCN) + t * SRCN + src] = bv;
  hTbf[((size_t)(b * SRCN + src)) * (25 * CC) + (24 - t) * CC + c] = bv;
}

// s *= (num+eps)/(den+eps); also refresh interior of the bf16 s-mirror.
__global__ void s_update_kernel(float* __restrict__ s, const float* __restrict__ num,
                                const float* __restrict__ den,
                                unsigned short* __restrict__ sbf, int n) {
  int i = blockIdx.x * blockDim.x + threadIdx.x;
  if (i >= n) return;
  float v = s[i] * (num[i] + EPSF) / (den[i] + EPSF);
  s[i] = v;
  int p = i % HWN;
  int img = i / HWN;
  int y = p / WW, x = p % WW;
  sbf[(size_t)img * MIRSZ + (size_t)(y + 2) * PW + (x + 2)] = f2bf(v);
}

// h[b][c][src][t] *= (num2[b][src][c][t]+eps)/(den2+eps)
__global__ void h_update_kernel(float* __restrict__ h, const float* __restrict__ num2,
                                const float* __restrict__ den2, int n) {
  int idx = blockIdx.x * blockDim.x + threadIdx.x;
  if (idx >= n) return;
  int t = idx % 25;
  int r = idx / 25;
  int src = r % SRCN; r /= SRCN;
  int c = r % CC;
  int b = r / CC;
  size_t j = (((size_t)(b * SRCN + src)) * CC + c) * 25 + t;
  h[idx] *= (num2[j] + EPSF) / (den2[j] + EPSF);
}

__global__ void zero_kernel(float* __restrict__ p, int n) {
  int i = blockIdx.x * blockDim.x + threadIdx.x;
  if (i < n) p[i] = 0.f;
}
__global__ void zero16_kernel(unsigned short* __restrict__ p, int n) {
  int i = blockIdx.x * blockDim.x + threadIdx.x;
  if (i < n) p[i] = 0;
}

// ---------------------------------------------------------------------------
// Per-sample 5x5 conv as implicit GEMM on v_wmma_f32_16x16x32_bf16.
//   out[b,o,y,x] = sum_{kk} wbf[b,o,kk] * in_mirror[b,i, y+ty, x0..],
//   kk = t*ICH + i. Input is the zero-halo bf16 mirror, so tile rows are
//   contiguous 144B segments -> staged with GLOBAL_LOAD_ASYNC_TO_LDS_B128.
// ---------------------------------------------------------------------------
template <int ICH, int OCH, bool BFOUT>
__global__ __launch_bounds__(128) void conv_wmma_kernel(
    const unsigned short* __restrict__ in,   // [b*ICH][PH][PW] bf16 mirror
    const unsigned short* __restrict__ wbf,  // [b*OCH][25*ICH] bf16
    float* __restrict__ outF,                // [b*OCH][H][W]   (if !BFOUT)
    unsigned short* __restrict__ outBf) {    // mirror          (if BFOUT)
  constexpr int KD = 25 * ICH;
  constexpr int NCH = (KD + 31) / 32;
  constexpr int TW = 64;
  constexpr int TILEW = 72;  // 68 used + pad to 144B rows
  __shared__ __attribute__((aligned(16))) unsigned short tile[ICH * 5 * TILEW];

  const int xt = blockIdx.x;
  const int y = blockIdx.y;
  const int z = blockIdx.z;
  const int ob = z % (OCH / 16);
  const int b = z / (OCH / 16);
  const int tid = threadIdx.x;
  const int lane = tid & 31;
  const int wave = tid >> 5;
  const int x0 = xt * TW;

#if HAS_ASYNC
  {
    constexpr int SEGS = TILEW / 8;  // 9 x 16B per tile row
    for (int idx = tid; idx < ICH * 5 * SEGS; idx += 128) {
      int seg = idx % SEGS;
      int row = idx / SEGS;
      int ry = row % 5;
      int i = row / 5;
      const unsigned short* g =
          in + ((size_t)(b * ICH + i) * PH + (y + ry)) * PW + x0 + seg * 8;
      async_cp16(g, &tile[row * TILEW + seg * 8]);
    }
    wait_async0();
  }
#else
  for (int idx = tid; idx < ICH * 5 * TILEW; idx += 128) {
    int cx = idx % TILEW;
    int row = idx / TILEW;
    int ry = row % 5;
    int i = row / 5;
    tile[idx] = in[((size_t)(b * ICH + i) * PH + (y + ry)) * PW + x0 + cx];
  }
#endif
  __syncthreads();

  const int m = lane & 15;
  const int khalf = lane >> 4;
  const unsigned short* wrow = wbf + (size_t)(b * OCH + ob * 16 + m) * KD;

  v8f acc = {};
  for (int kc = 0; kc < NCH; ++kc) {
    const int kbase = kc * 32;
    v16u aU, bU;
#pragma unroll
    for (int j = 0; j < 16; ++j) {  // A: 16x32 bf16, lane = M row
      int kj = j + khalf * 16;
      int kk = kbase + kj;
      aU[j] = (kk < KD) ? wrow[kk] : (unsigned short)0;
    }
    // B: 32x16 bf16. ICH is a power of two and k0 is 16-aligned, so tap (t)
    // is constant across the 16 packed elements -> immediate-offset ds loads.
    const int k0 = kbase + khalf * 16;
    if (k0 < KD) {
      const int t = k0 / ICH;
      const int i0 = k0 - t * ICH;
      const int ty = t / 5;
      const int tx = t - 5 * ty;
      const int base = (i0 * 5 + ty) * TILEW + wave * 16 + m + tx;
#pragma unroll
      for (int j = 0; j < 16; ++j) bU[j] = tile[base + j * 5 * TILEW];
    } else {
#pragma unroll
      for (int j = 0; j < 16; ++j) bU[j] = 0;
    }
    acc = wmma_bf16(aU, bU, acc);
  }

  const int px = x0 + wave * 16 + m;
#pragma unroll
  for (int r = 0; r < 8; ++r) {
    int mo = r + 8 * khalf;
    if constexpr (BFOUT)
      outBf[(size_t)(b * OCH + ob * 16 + mo) * MIRSZ + (size_t)(y + 2) * PW + (px + 2)] =
          f2bf(acc[r]);
    else
      outF[(((size_t)b * OCH + ob * 16 + mo) * HH + y) * WW + px] = acc[r];
  }
}

// ---------------------------------------------------------------------------
// Full-image correlation: out25[b,src,c,t] += sum_{y,x} s[src,y+ty-2,x+tx-2]*xm[c,y,x]
// Both operands come from zero-halo mirrors (no bounds checks). Waves stride
// the 25 taps; K = pixels in 32-chunks; merged via global_atomic_add_f32.
// ---------------------------------------------------------------------------
#define RROWS 2
__global__ __launch_bounds__(128) void sconv_wmma_kernel(
    const unsigned short* __restrict__ sPad,  // [B*16][PH][PW]
    const unsigned short* __restrict__ xPad,  // [B*64][PH][PW]
    float* __restrict__ out25) {              // [B,16,64,25], pre-zeroed
  __shared__ __attribute__((aligned(16))) unsigned short xs[16 * RROWS * PW];
  __shared__ __attribute__((aligned(16))) unsigned short ss[16 * (RROWS + 4) * PW];
  const int y0 = blockIdx.x * RROWS;
  const int cb = blockIdx.y;
  const int b = blockIdx.z;
  const int tid = threadIdx.x;
  const int lane = tid & 31;
  const int wave = tid >> 5;

#if HAS_ASYNC
  {
    constexpr int SEG = PW / 8;  // 33 x 16B per mirror row
    for (int idx = tid; idx < 16 * RROWS * SEG; idx += 128) {
      int seg = idx % SEG;
      int row = idx / SEG;
      int r = row % RROWS;
      int i = row / RROWS;
      const unsigned short* g =
          xPad + ((size_t)(b * CC + cb * 16 + i) * PH + (y0 + 2 + r)) * PW + seg * 8;
      async_cp16(g, &xs[(i * RROWS + r) * PW + seg * 8]);
    }
    for (int idx = tid; idx < 16 * (RROWS + 4) * SEG; idx += 128) {
      int seg = idx % SEG;
      int row = idx / SEG;
      int r = row % (RROWS + 4);
      int i = row / (RROWS + 4);
      const unsigned short* g =
          sPad + ((size_t)(b * SRCN + i) * PH + (y0 + r)) * PW + seg * 8;
      async_cp16(g, &ss[(i * (RROWS + 4) + r) * PW + seg * 8]);
    }
    wait_async0();
  }
#else
  for (int idx = tid; idx < 16 * RROWS * PW; idx += 128) {
    int xx = idx % PW;
    int row = idx / PW;
    int r = row % RROWS;
    int i = row / RROWS;
    xs[idx] = xPad[((size_t)(b * CC + cb * 16 + i) * PH + (y0 + 2 + r)) * PW + xx];
  }
  for (int idx = tid; idx < 16 * (RROWS + 4) * PW; idx += 128) {
    int xx = idx % PW;
    int row = idx / PW;
    int r = row % (RROWS + 4);
    int i = row / (RROWS + 4);
    ss[idx] = sPad[((size_t)(b * SRCN + i) * PH + (y0 + r)) * PW + xx];
  }
#endif
  __syncthreads();

  const v8f zv = {};
  v8f acc[7];
#pragma unroll
  for (int g = 0; g < 7; ++g) acc[g] = zv;
  const int m = lane & 15;
  const int khalf = lane >> 4;

  for (int yr = 0; yr < RROWS; ++yr) {
    for (int pc = 0; pc < WW / 32; ++pc) {
      const int pxb = pc * 32;
      v16u bU;  // B: 32 px x 16 c from x row (shared by all taps)
      const int xrow = (m * RROWS + yr) * PW + pxb + 2 + khalf * 16;
#pragma unroll
      for (int j = 0; j < 16; ++j) bU[j] = xs[xrow + j];
#pragma unroll
      for (int g = 0; g < 7; ++g) {
        int t = wave + 4 * g;
        if (t < 25) {
          int ty = t / 5, tx = t - 5 * (t / 5);
          v16u aU;  // A: 16 src x 32 shifted-s pixels (halo makes shifts safe)
          const int sbase = (m * (RROWS + 4) + yr + ty) * PW + pxb + tx;
#pragma unroll
          for (int j = 0; j < 16; ++j) {
            int kj = j + khalf * 16;
            aU[j] = ss[sbase + kj];
          }
          acc[g] = wmma_bf16(aU, bU, acc[g]);
        }
      }
    }
  }

  const int c = cb * 16 + m;
#pragma unroll
  for (int g = 0; g < 7; ++g) {
    int t = wave + 4 * g;
    if (t < 25) {
#pragma unroll
      for (int r = 0; r < 8; ++r) {
        int src = r + 8 * khalf;
        atomicAdd(&out25[(((size_t)b * SRCN + src) * CC + c) * 25 + t], acc[g][r]);
      }
    }
  }
}

// ---------------------------------------------------------------------------
extern "C" void kernel_launch(void* const* d_in, const int* in_sizes, int n_in,
                              void* d_out, int out_size, void* d_ws, size_t ws_size,
                              hipStream_t stream) {
  (void)in_sizes; (void)n_in; (void)out_size; (void)ws_size;
  const float* x = (const float*)d_in[0];      // [B,64,H,W]
  const float* h0 = (const float*)d_in[1];     // [64,16,5,5]
  const float* W_lin = (const float*)d_in[2];  // [16,64]
  const float* b_lin = (const float*)d_in[3];  // [16]
  float* out = (float*)d_out;                  // [B,64,H,W]

  const size_t NS = (size_t)BB * SRCN * HWN;  // 8,388,608
  const int NH = BB * CC * SRCN * 25;         // 204,800
  const int NMIRX = BB * CC;                  // 512 images
  const int NMIRS = BB * SRCN;                // 128 images

  float* s = (float*)d_ws;                      // NS f32
  float* h = s + NS;                            // NH f32
  float* o25A = h + NH;                         // NH f32
  float* o25B = o25A + NH;                      // NH f32
  unsigned short* hbf = (unsigned short*)(o25B + NH);
  unsigned short* hTbf = hbf + NH;
  unsigned short* xbf = hTbf + NH;              // NMIRX*MIRSZ bf16
  unsigned short* sbf = xbf + (size_t)NMIRX * MIRSZ;
  unsigned short* cbf = sbf + (size_t)NMIRS * MIRSZ;
  // num/den live inside d_out (2*NS = 16.7M floats <= 33.5M); final conv
  // overwrites d_out after their last use.
  float* numb = out;
  float* denb = numb + NS;

  const dim3 convGrid16(WW / 64, HH, BB * (SRCN / 16));  // O=16
  const dim3 convGrid64(WW / 64, HH, BB * (CC / 16));    // O=64
  const dim3 sconvGrid(HH / RROWS, CC / 16, BB);

  pack_pad_kernel<<<(NMIRX * MIRSZ + 255) / 256, 256, 0, stream>>>(x, xbf, NMIRX);
  init_s_kernel<<<(BB * HWN + 255) / 256, 256, 0, stream>>>(x, W_lin, b_lin, s);
  pack_pad_kernel<<<(NMIRS * MIRSZ + 255) / 256, 256, 0, stream>>>(s, sbf, NMIRS);
  zero16_kernel<<<(NMIRX * MIRSZ + 255) / 256, 256, 0, stream>>>(cbf, NMIRX * MIRSZ);
  h_init_kernel<<<(NH + 255) / 256, 256, 0, stream>>>(h0, h, NH);

  for (int it = 0; it < 2; ++it) {
    prep_weights_kernel<<<(NH + 255) / 256, 256, 0, stream>>>(h, hbf, hTbf, NH);
    // num = conv(x, hT); cbf = conv(s,h); den = conv(cbf, hT)
    conv_wmma_kernel<CC, SRCN, false><<<convGrid16, 128, 0, stream>>>(xbf, hTbf, numb, nullptr);
    conv_wmma_kernel<SRCN, CC, true><<<convGrid64, 128, 0, stream>>>(sbf, hbf, nullptr, cbf);
    conv_wmma_kernel<CC, SRCN, false><<<convGrid16, 128, 0, stream>>>(cbf, hTbf, denb, nullptr);
    s_update_kernel<<<((int)NS + 255) / 256, 256, 0, stream>>>(s, numb, denb, sbf, (int)NS);
    // num2 = sconv(s, x); den2 = sconv(s, conv(s,h))
    zero_kernel<<<(2 * NH + 255) / 256, 256, 0, stream>>>(o25A, 2 * NH);
    sconv_wmma_kernel<<<sconvGrid, 128, 0, stream>>>(sbf, xbf, o25A);
    conv_wmma_kernel<SRCN, CC, true><<<convGrid64, 128, 0, stream>>>(sbf, hbf, nullptr, cbf);
    sconv_wmma_kernel<<<sconvGrid, 128, 0, stream>>>(sbf, cbf, o25B);
    h_update_kernel<<<(NH + 255) / 256, 256, 0, stream>>>(h, o25A, o25B, NH);
  }

  prep_weights_kernel<<<(NH + 255) / 256, 256, 0, stream>>>(h, hbf, hTbf, NH);
  conv_wmma_kernel<SRCN, CC, false><<<convGrid64, 128, 0, stream>>>(sbf, hbf, out, nullptr);
}
